// AdaptiveMixtureOfExperts_25795573580557
// MI455X (gfx1250) — matrix operand order repaired and checked by reading, not compiled, not run on hardware
//
#include <hip/hip_runtime.h>
#include <hip/hip_bf16.h>

#define NUM_EXPERTS 8
#define INPUT_DIM   1024
#define TOKENS      1024
#define HIDDEN      512
#define MAXH        4096

typedef __attribute__((ext_vector_type(16))) __bf16 v16bf;
typedef __attribute__((ext_vector_type(8)))  float  v8f;

struct __align__(16) U2 { unsigned long long x, y; };
union Frag { v16bf v; U2 q[2]; };

__device__ __forceinline__ float gelu_exact(float x) {
  return 0.5f * x * (1.0f + erff(x * 0.7071067811865476f));
}

__global__ __launch_bounds__(256) void zero_f32(float* p, int n) {
  int i = blockIdx.x * 256 + threadIdx.x;
  if (i < n) p[i] = 0.0f;
}
__global__ void zero_i32(int* p, int n) {
  int i = threadIdx.x;
  if (i < n) p[i] = 0;
}

// Build bf16 x and bf16 router input [x | broadcast(liquid)].
__global__ __launch_bounds__(256) void build_inputs(const float* __restrict__ x,
                                                    const float* __restrict__ liq,
                                                    __bf16* __restrict__ xbf,
                                                    __bf16* __restrict__ ribf) {
  int i = blockIdx.x * 256 + threadIdx.x;
  if (i >= TOKENS * 1280) return;
  int t = i / 1280, j = i - t * 1280;
  float v;
  if (j < INPUT_DIM) { v = x[t * INPUT_DIM + j]; xbf[t * INPUT_DIM + j] = (__bf16)v; }
  else               { v = liq[(t >> 9) * 256 + (j - INPUT_DIM)]; }
  ribf[i] = (__bf16)v;
}

// W [K,N] fp32 row-major  ->  Wt [N,K] bf16 row-major (coalesced via LDS tile)
__global__ __launch_bounds__(256) void transpose_bf16(const float* __restrict__ W,
                                                      __bf16* __restrict__ Wt,
                                                      int K, int N) {
  __shared__ float tile[32][33];
  int n0 = blockIdx.x * 32, k0 = blockIdx.y * 32;
  int tx = threadIdx.x, ty = threadIdx.y; // block (32,8)
  #pragma unroll
  for (int i = 0; i < 32; i += 8) {
    int k = k0 + ty + i, n = n0 + tx;
    if (k < K && n < N) tile[ty + i][tx] = W[(size_t)k * N + n];
  }
  __syncthreads();
  #pragma unroll
  for (int i = 0; i < 32; i += 8) {
    int n = n0 + ty + i, k = k0 + tx;
    if (n < N && k < K) Wt[(size_t)n * K + k] = (__bf16)tile[tx][ty + i];
  }
}

// C = act(A[M,K] @ Bt[N,K]^T + bias)
// Block tile 128x128, 8 waves, wave tile 32x64 (2x4 WMMA frags), K step 32,
// double-buffered LDS (one barrier per K step). Global loads for the next
// step are issued right before their ds_store (short register lifetime, no
// spills); latency is covered by global_prefetch of the K+2 tile.
// MODE 0: plain rows; out bf16 = gelu(acc+bias)                   (router/unc layer 1)
// MODE 1: A rows gathered via idx; out fp32 = acc+bias            (expert layer 1)
// MODE 2: A rows compacted; out[idx[r],:] += gate * (acc+bias)    (expert layer 2)
template <int MODE>
__global__ __launch_bounds__(256) void gemm_bf16(
    const __bf16* __restrict__ A, int lda,
    const __bf16* __restrict__ Bt, int K,
    const float* __restrict__ bias,
    void* __restrict__ Cout, int ldo,
    int Mfull, const int* __restrict__ cntp,
    const int* __restrict__ idx,
    const float* __restrict__ gate, int eidx) {
  const int cnt = (MODE == 0) ? Mfull : *cntp;
  const int bm = blockIdx.y, bn = blockIdx.x;
  if (bm * 128 >= cnt) return;

  __shared__ __align__(16) __bf16 As[2][128 * 40];
  __shared__ __align__(16) __bf16 Bs[2][128 * 40];

  const int tid  = threadIdx.x;
  const int lane = tid & 31;
  const int wave = tid >> 5;
  const int wm = (wave & 3) * 32;   // wave tile row origin (4 waves in M)
  const int wn = (wave >> 2) * 64;  // wave tile col origin (2 waves in N)

  v8f acc[2][4];
  #pragma unroll
  for (int i = 0; i < 2; ++i)
    #pragma unroll
    for (int j = 0; j < 4; ++j)
      #pragma unroll
      for (int k = 0; k < 8; ++k) acc[i][j][k] = 0.0f;

  // Global->LDS assignments (rows fixed across K loop). A,B: 2x 16B chunks/thread.
  const __bf16* aptr[2];
  const __bf16* bptr[2];
  bool aval[2];
  int  alds[2];
  #pragma unroll
  for (int i = 0; i < 2; ++i) {
    int r = (tid >> 2) + i * 64;             // 0..127
    int grow = bm * 128 + r;
    bool v; int arow;
    if (MODE == 0)      { v = true;       arow = grow; }   // full-M launches
    else if (MODE == 1) { v = grow < cnt; arow = v ? idx[grow] : 0; }
    else                { v = grow < cnt; arow = v ? grow : 0; }
    aval[i] = v;
    aptr[i] = A + (size_t)arow * lda + (tid & 3) * 8;
    alds[i] = r * 40 + (tid & 3) * 8;
    bptr[i] = Bt + (size_t)(bn * 128 + r) * K + (tid & 3) * 8;
  }

  const int lr = lane & 15;
  const int ah = (lane >> 4) * 8;   // A frag K-offset (ISA: lanes16-31 start at K=8)
  const int bh = (lane >> 4) * 8;   // B frag K-offset (lanes16-31 start at K=8)

  const int nsteps = K >> 5;

  // Prologue: stage K-step 0 into buffer 0.
  {
    #pragma unroll
    for (int i = 0; i < 2; ++i) {
      U2 a; a.x = 0ULL; a.y = 0ULL;
      if (aval[i]) a = *(const U2*)(aptr[i]);
      U2 b = *(const U2*)(bptr[i]);
      *(U2*)(&As[0][alds[i]]) = a;
      *(U2*)(&Bs[0][alds[i]]) = b;
    }
    __syncthreads();
  }

  for (int it = 0; it < nsteps; ++it) {
    const int cur = it & 1;
    const bool more = (it + 1) < nsteps;

    if (it + 2 < nsteps) {
      const int kp = (it + 2) << 5;
      __builtin_prefetch(aptr[0] + kp, 0, 0);
      __builtin_prefetch(aptr[1] + kp, 0, 0);
      __builtin_prefetch(bptr[0] + kp, 0, 0);
      __builtin_prefetch(bptr[1] + kp, 0, 0);
    }

    // Compute from LDS buffer `cur`.
    Frag fa[2], fb[4];
    #pragma unroll
    for (int j = 0; j < 4; ++j) {
      const __bf16* q = &Bs[cur][(wn + j * 16 + lr) * 40 + bh];
      fb[j].q[0] = *(const U2*)q;            // K = bh..bh+7
      fb[j].q[1] = *(const U2*)(q + 16);     // K = bh+16..bh+23
    }
    #pragma unroll
    for (int i = 0; i < 2; ++i) {
      const __bf16* p = &As[cur][(wm + i * 16 + lr) * 40 + ah];
      fa[i].q[0] = *(const U2*)p;            // K = ah..ah+7
      fa[i].q[1] = *(const U2*)(p + 16);     // K = ah+16..ah+23
    }
    #pragma unroll
    for (int i = 0; i < 2; ++i)
      #pragma unroll
      for (int j = 0; j < 4; ++j)
        acc[i][j] = __builtin_amdgcn_wmma_f32_16x16x32_bf16(
            false, fa[i].v, false, fb[j].v, (short)0, acc[i][j], false, false);

    if (more) {
      // Load next tile and immediately forward it to LDS: staging registers
      // live only a few instructions (prefetch above makes these cache hits).
      const int k0 = (it + 1) << 5;
      const int nxt = cur ^ 1;
      #pragma unroll
      for (int i = 0; i < 2; ++i) {
        U2 a; a.x = 0ULL; a.y = 0ULL;
        if (aval[i]) a = *(const U2*)(aptr[i] + k0);
        U2 b = *(const U2*)(bptr[i] + k0);
        *(U2*)(&As[nxt][alds[i]]) = a;
        *(U2*)(&Bs[nxt][alds[i]]) = b;
      }
      __syncthreads();
    }
  }

  #pragma unroll
  for (int i = 0; i < 2; ++i)
    #pragma unroll
    for (int j = 0; j < 4; ++j)
      #pragma unroll
      for (int v = 0; v < 8; ++v) {
        int grow = bm * 128 + wm + i * 16 + (lane >> 4) * 8 + v;
        int gcol = bn * 128 + wn + j * 16 + (lane & 15);
        float val = acc[i][j][v];
        if (MODE == 0) {
          if (grow < cnt)
            ((__bf16*)Cout)[(size_t)grow * ldo + gcol] =
                (__bf16)gelu_exact(val + bias[gcol]);
        } else if (MODE == 1) {
          if (grow < cnt)
            ((float*)Cout)[(size_t)grow * ldo + gcol] = val + bias[gcol];
        } else {
          if (grow < cnt) {
            int t = idx[grow];
            float g = gate[t * 8 + eidx];
            float* o = (float*)Cout + (size_t)t * INPUT_DIM + gcol;
            *o += g * (val + bias[gcol]);
          }
        }
      }
}

// One wave per token: router logits + softmax + top2 gates + compaction + uncertainty.
__global__ __launch_bounds__(256) void router_finalize(
    const __bf16* __restrict__ hrb, const float* __restrict__ Wr2, const float* __restrict__ br2,
    const __bf16* __restrict__ hub, const float* __restrict__ Wu2, const float* __restrict__ bu2,
    float* __restrict__ gates, int* __restrict__ cnt, int* __restrict__ idxbuf,
    float* __restrict__ uarr) {
  const int lane = threadIdx.x & 31;
  const int t = blockIdx.x * 8 + (threadIdx.x >> 5);
  float acc[8];
  #pragma unroll
  for (int e = 0; e < 8; ++e) acc[e] = 0.0f;
  for (int k = lane; k < 512; k += 32) {
    float h = (float)hrb[t * 512 + k];
    #pragma unroll
    for (int e = 0; e < 8; ++e) acc[e] += h * Wr2[k * 8 + e];
  }
  float ua = 0.0f;
  for (int k = lane; k < 256; k += 32) ua += (float)hub[t * 256 + k] * Wu2[k];
  #pragma unroll
  for (int e = 0; e < 8; ++e)
    for (int off = 16; off > 0; off >>= 1) acc[e] += __shfl_down(acc[e], off, 32);
  for (int off = 16; off > 0; off >>= 1) ua += __shfl_down(ua, off, 32);
  if (lane == 0) {
    float l[8], m = -3.0e38f;
    #pragma unroll
    for (int e = 0; e < 8; ++e) { l[e] = acc[e] + br2[e]; m = fmaxf(m, l[e]); }
    float s = 0.0f;
    #pragma unroll
    for (int e = 0; e < 8; ++e) { l[e] = expf(l[e] - m); s += l[e]; }
    float inv = 1.0f / s;
    #pragma unroll
    for (int e = 0; e < 8; ++e) l[e] *= inv;
    int i1 = 0; float p1 = l[0];
    #pragma unroll
    for (int e = 1; e < 8; ++e) if (l[e] > p1) { p1 = l[e]; i1 = e; }  // lowest idx wins ties
    int i2 = -1; float p2 = -1.0f;
    #pragma unroll
    for (int e = 0; e < 8; ++e) if (e != i1 && l[e] > p2) { p2 = l[e]; i2 = e; }
    float gs = 1.0f / (p1 + p2);
    gates[t * 8 + i1] = p1 * gs;
    gates[t * 8 + i2] = p2 * gs;
    int s1 = atomicAdd(&cnt[i1], 1); idxbuf[i1 * TOKENS + s1] = t;
    int s2 = atomicAdd(&cnt[i2], 1); idxbuf[i2 * TOKENS + s2] = t;
    uarr[t] = 1.0f / (1.0f + expf(-(ua + bu2[0])));
  }
}

// Row-wise LayerNorm + exact GELU, fp32 -> bf16, for rows < cnt.
__global__ __launch_bounds__(256) void ln_gelu(const float* __restrict__ H,
                                               const float* __restrict__ gma,
                                               const float* __restrict__ bta,
                                               __bf16* __restrict__ Aout,
                                               const int* __restrict__ cntp, int h) {
  const int r = blockIdx.x;
  if (r >= *cntp) return;
  __shared__ float s1[256], s2[256];
  const int tid = threadIdx.x;
  float sum = 0.0f, sq = 0.0f;
  for (int k = tid; k < h; k += 256) {
    float v = H[(size_t)r * h + k];
    sum += v; sq += v * v;
  }
  s1[tid] = sum; s2[tid] = sq;
  __syncthreads();
  for (int o = 128; o > 0; o >>= 1) {
    if (tid < o) { s1[tid] += s1[tid + o]; s2[tid] += s2[tid + o]; }
    __syncthreads();
  }
  float mu = s1[0] / (float)h;
  float var = s2[0] / (float)h - mu * mu;
  float rstd = rsqrtf(var + 1e-5f);
  for (int k = tid; k < h; k += 256) {
    float v = (H[(size_t)r * h + k] - mu) * rstd * gma[k] + bta[k];
    Aout[(size_t)r * h + k] = (__bf16)gelu_exact(v);
  }
}

// Deterministic tail: mean uncertainty, expert loads, load-balance loss.
__global__ __launch_bounds__(256) void tail_kernel(const float* __restrict__ uarr,
                                                   const int* __restrict__ cnt,
                                                   float* __restrict__ out_tail) {
  __shared__ float s[256];
  const int tid = threadIdx.x;
  float x = 0.0f;
  for (int t = tid; t < TOKENS; t += 256) x += uarr[t];
  s[tid] = x;
  __syncthreads();
  for (int o = 128; o > 0; o >>= 1) { if (tid < o) s[tid] += s[tid + o]; __syncthreads(); }
  if (tid == 0) {
    float total = 0.0f;
    for (int e = 0; e < 8; ++e) total += (float)cnt[e];
    float lb = 0.0f;
    for (int e = 0; e < 8; ++e) {
      float le = (float)cnt[e] / total;
      out_tail[1 + e] = le;
      float d = le - 0.125f;
      lb += d * d;
    }
    out_tail[0] = 0.01f * lb / 8.0f;
    out_tail[9] = s[0] / (float)TOKENS;
  }
}

extern "C" void kernel_launch(void* const* d_in, const int* in_sizes, int n_in,
                              void* d_out, int out_size, void* d_ws, size_t ws_size,
                              hipStream_t stream) {
  (void)in_sizes; (void)n_in; (void)out_size; (void)ws_size;
  const float* x   = (const float*)d_in[0];
  const float* liq = (const float*)d_in[1];
  const float* Wr1 = (const float*)d_in[2];
  const float* br1 = (const float*)d_in[3];
  const float* Wr2 = (const float*)d_in[4];
  const float* br2 = (const float*)d_in[5];
  const float* Wu1 = (const float*)d_in[6];
  const float* bu1 = (const float*)d_in[7];
  const float* Wu2 = (const float*)d_in[8];
  const float* bu2 = (const float*)d_in[9];

  float* out = (float*)d_out;
  float* out_tail = out + (size_t)TOKENS * INPUT_DIM;

  char* w = (char*)d_ws;
  auto carve = [&](size_t bytes) { char* p = w; w += (bytes + 255) & ~(size_t)255; return p; };
  __bf16* xbf  = (__bf16*)carve((size_t)TOKENS * 1024 * 2);
  __bf16* ribf = (__bf16*)carve((size_t)TOKENS * 1280 * 2);
  __bf16* hrb  = (__bf16*)carve((size_t)TOKENS * 512 * 2);
  __bf16* hub  = (__bf16*)carve((size_t)TOKENS * 256 * 2);
  __bf16* W1t  = (__bf16*)carve((size_t)MAXH * 1280 * 2);   // also holds Wr1t / Wu1t
  __bf16* W2t  = (__bf16*)carve((size_t)1024 * MAXH * 2);
  float*  Hbuf = (float*) carve((size_t)TOKENS * MAXH * 4);
  __bf16* Abf  = (__bf16*)carve((size_t)TOKENS * MAXH * 2);
  float*  gates= (float*) carve((size_t)TOKENS * 8 * 4);
  float*  uarr = (float*) carve((size_t)TOKENS * 4);
  int*    cnt  = (int*)   carve(8 * 4);
  int*    idx  = (int*)   carve((size_t)8 * TOKENS * 4);

  zero_f32<<<(TOKENS * INPUT_DIM + 255) / 256, 256, 0, stream>>>(out, TOKENS * INPUT_DIM);
  zero_f32<<<(TOKENS * 8 + 255) / 256, 256, 0, stream>>>(gates, TOKENS * 8);
  zero_i32<<<1, 32, 0, stream>>>(cnt, 8);

  build_inputs<<<(TOKENS * 1280 + 255) / 256, 256, 0, stream>>>(x, liq, xbf, ribf);

  // Router layer 1: [1024,1280] @ [1280,512], bias+GELU -> bf16
  transpose_bf16<<<dim3(512 / 32, 1280 / 32), dim3(32, 8), 0, stream>>>(Wr1, W1t, 1280, 512);
  gemm_bf16<0><<<dim3(512 / 128, TOKENS / 128), 256, 0, stream>>>(
      ribf, 1280, W1t, 1280, br1, hrb, 512, TOKENS, nullptr, nullptr, nullptr, 0);

  // Uncertainty layer 1: [1024,1024] @ [1024,256], bias+GELU -> bf16
  transpose_bf16<<<dim3(256 / 32, 1024 / 32), dim3(32, 8), 0, stream>>>(Wu1, W1t, 1024, 256);
  gemm_bf16<0><<<dim3(256 / 128, TOKENS / 128), 256, 0, stream>>>(
      xbf, 1024, W1t, 1024, bu1, hub, 256, TOKENS, nullptr, nullptr, nullptr, 0);

  router_finalize<<<TOKENS / 8, 256, 0, stream>>>(hrb, Wr2, br2, hub, Wu2, bu2,
                                                  gates, cnt, idx, uarr);

  for (int e = 0; e < NUM_EXPERTS; ++e) {
    const int h = HIDDEN * (e + 1);
    const float* W1 = (const float*)d_in[10 + e * 6 + 0];
    const float* b1 = (const float*)d_in[10 + e * 6 + 1];
    const float* gm = (const float*)d_in[10 + e * 6 + 2];
    const float* bt = (const float*)d_in[10 + e * 6 + 3];
    const float* W2 = (const float*)d_in[10 + e * 6 + 4];
    const float* b2 = (const float*)d_in[10 + e * 6 + 5];

    transpose_bf16<<<dim3(h / 32, 1024 / 32), dim3(32, 8), 0, stream>>>(W1, W1t, 1024, h);
    transpose_bf16<<<dim3(1024 / 32, h / 32), dim3(32, 8), 0, stream>>>(W2, W2t, h, 1024);

    gemm_bf16<1><<<dim3(h / 128, TOKENS / 128), 256, 0, stream>>>(
        xbf, 1024, W1t, 1024, b1, Hbuf, h, TOKENS, cnt + e, idx + e * TOKENS, nullptr, 0);

    ln_gelu<<<TOKENS, 256, 0, stream>>>(Hbuf, gm, bt, Abf, cnt + e, h);

    gemm_bf16<2><<<dim3(1024 / 128, TOKENS / 128), 256, 0, stream>>>(
        Abf, h, W2t, h, b2, out, 1024, TOKENS, cnt + e, idx + e * TOKENS, gates, e);
  }

  tail_kernel<<<1, 256, 0, stream>>>(uarr, cnt, out_tail);
}
